// AttentionDiffusion_13073880449658
// MI455X (gfx1250) — matrix operand order repaired and checked by reading, not compiled, not run on hardware
//
#include <hip/hip_runtime.h>
#include <hip/hip_bf16.h>
#include <math.h>

#define TD 128
#define EPSV 1e-6f
#define CLIP_HI 1e6f

typedef __attribute__((ext_vector_type(16))) _Float16 v16h;
typedef __attribute__((ext_vector_type(8)))  float    v8f;

// ---------------------------------------------------------------- zero gsum
__global__ void k_zero128(float* p) { p[threadIdx.x] = 0.0f; }

// ------------------------------------------------- column means -> g[128]
// g = 0.5*(mean(xA,axis=0) + mean(xB,axis=0)).
// stride is a multiple of 128, so each thread's column index is invariant.
__global__ void k_colsum(const float* __restrict__ xA, const float* __restrict__ xB,
                         int NA, int NB, float* __restrict__ gsum) {
  const long long totalA = (long long)NA * TD;
  const long long totalB = (long long)NB * TD;
  const long long stride = (long long)gridDim.x * blockDim.x;
  const long long i0 = (long long)blockIdx.x * blockDim.x + threadIdx.x;
  float sA = 0.0f, sB = 0.0f;
  for (long long i = i0; i < totalA; i += stride) sA += xA[i];
  for (long long i = i0; i < totalB; i += stride) sB += xB[i];
  const int col = (int)(i0 % TD);
  atomicAdd(&gsum[col], sA * (0.5f / (float)NA) + sB * (0.5f / (float)NB));
}

// ------------------------------------------------- alpha via WMMA matvec
// h = tanh(fc1_w @ g + fc1_b); alpha = clamp(sigmoid(dot(h, fc2_w) + fc2_b)).
// 8 waves, wave r computes rows [16r, 16r+16). A = fc1_w tile (16x32 f16),
// B = g broadcast across all 16 columns (32x16 f16) -> every D column holds
// the matvec partial. Accumulate 4 k-tiles of 32 in f32.
__global__ void k_alpha(const float* __restrict__ gsum,
                        const float* __restrict__ fc1_w, const float* __restrict__ fc1_b,
                        const float* __restrict__ fc2_w, const float* __restrict__ fc2_b,
                        float* __restrict__ alpha_out) {
  __shared__ float sg[TD];
  __shared__ float sh[TD];
  const int tid = threadIdx.x;            // 256 threads = 8 waves (wave32)
  if (tid < TD) sg[tid] = gsum[tid];
  __syncthreads();

  const int wave  = tid >> 5;
  const int lane  = tid & 31;
  const int mrow  = (wave << 4) + (lane & 15);      // A row this lane carries
  const int kbase = (lane < 16) ? 0 : 8;            // A-layout half-wave K base

  v8f c = {};
  for (int kt = 0; kt < 4; ++kt) {
    v16h a, b;
    const float* wrow = fc1_w + (size_t)mrow * TD + kt * 32;
#pragma unroll
    for (int j = 0; j < 8; ++j) {
      a[j]     = (_Float16)wrow[kbase + j];          // K = kbase..kbase+7
      a[8 + j] = (_Float16)wrow[16 + kbase + j];     // K = 16+kbase..
    }
    // B 32x16 layout: lanes 0-15 hold K=0..7,16..23; lanes 16-31 hold K=8..15,24..31.
    const float* gk = sg + kt * 32 + ((lane < 16) ? 0 : 8);
#pragma unroll
    for (int j = 0; j < 8; ++j) { b[j] = (_Float16)gk[j]; b[8 + j] = (_Float16)gk[16 + j]; }
    c = __builtin_amdgcn_wmma_f32_16x16x32_f16(false, a, false, b,
                                               (short)0, c, false, false);
  }
  // D layout: lane 0 holds M=0..7 in c[0..7]; lane 16 holds M=8..15.
  if (lane == 0) {
#pragma unroll
    for (int v = 0; v < 8; ++v) {
      const int m = (wave << 4) + v;
      sh[m] = tanhf(c[v] + fc1_b[m]);
    }
  } else if (lane == 16) {
#pragma unroll
    for (int v = 0; v < 8; ++v) {
      const int m = (wave << 4) + 8 + v;
      sh[m] = tanhf(c[v] + fc1_b[m]);
    }
  }
  __syncthreads();
  if (tid == 0) {
    float acc = fc2_b[0];
    for (int j = 0; j < TD; ++j) acc += sh[j] * fc2_w[j];
    float alpha = 1.0f / (1.0f + expf(-acc));
    alpha = fminf(fmaxf(alpha, EPSV), 1.0f - EPSV);
    *alpha_out = alpha;
  }
}

// ------------------------------------------------- cur = x0 ; acc = x0*alpha
__global__ void k_prep(const float* __restrict__ x0, float* __restrict__ cur,
                       float* __restrict__ acc, const float* __restrict__ alpha_p,
                       long long n) {
  const float alpha = *alpha_p;
  const long long stride = (long long)gridDim.x * blockDim.x;
  for (long long i = (long long)blockIdx.x * blockDim.x + threadIdx.x; i < n; i += stride) {
    const float v = x0[i];
    cur[i] = v;
    acc[i] = v * alpha;
  }
}

// ------------------------------------------------- edge scatter-add
// One wave32 per edge: 32 lanes x float4 = one 128-float row.
// acc_dst[dst] += (1-alpha)*att[e] * cur_src[src]
__global__ void k_scatter(const float* __restrict__ cur_src, float* __restrict__ acc_dst,
                          const int* __restrict__ eidx, const float* __restrict__ att,
                          const float* __restrict__ alpha_p, int E) {
  const int e    = ((int)blockIdx.x << 3) + ((int)threadIdx.x >> 5);  // 8 waves/block
  const int lane = threadIdx.x & 31;
  if (e >= E) return;
  const int src = eidx[e];
  const int dst = eidx[E + e];
  const float w = (1.0f - *alpha_p) * att[e];
  const float4 v = *(const float4*)(cur_src + (size_t)src * TD + lane * 4);
  float* dp = acc_dst + (size_t)dst * TD + lane * 4;
  atomicAdd(dp + 0, w * v.x);   // no-return global_atomic_add_f32
  atomicAdd(dp + 1, w * v.y);
  atomicAdd(dp + 2, w * v.z);
  atomicAdd(dp + 3, w * v.w);
}

// ------------------------------------------------- cur = clip(acc); acc = x0*alpha
__global__ void k_advance(const float* __restrict__ x0, float* __restrict__ cur,
                          float* __restrict__ acc, const float* __restrict__ alpha_p,
                          long long n) {
  const float alpha = *alpha_p;
  const long long stride = (long long)gridDim.x * blockDim.x;
  for (long long i = (long long)blockIdx.x * blockDim.x + threadIdx.x; i < n; i += stride) {
    const float a = acc[i];
    cur[i] = fminf(fmaxf(a, EPSV), CLIP_HI);
    acc[i] = x0[i] * alpha;
  }
}

// ------------------------------------------------- out = clip(acc)
__global__ void k_final(const float* __restrict__ acc, float* __restrict__ out, long long n) {
  const long long stride = (long long)gridDim.x * blockDim.x;
  for (long long i = (long long)blockIdx.x * blockDim.x + threadIdx.x; i < n; i += stride) {
    out[i] = fminf(fmaxf(acc[i], EPSV), CLIP_HI);
  }
}

extern "C" void kernel_launch(void* const* d_in, const int* in_sizes, int n_in,
                              void* d_out, int out_size, void* d_ws, size_t ws_size,
                              hipStream_t stream) {
  (void)n_in; (void)out_size; (void)ws_size;
  const float* xA    = (const float*)d_in[0];
  const float* xB    = (const float*)d_in[1];
  const float* attAB = (const float*)d_in[2];
  const float* attBA = (const float*)d_in[3];
  const float* fc1_w = (const float*)d_in[4];
  const float* fc1_b = (const float*)d_in[5];
  const float* fc2_w = (const float*)d_in[6];
  const float* fc2_b = (const float*)d_in[7];
  const int*   eAB   = (const int*)d_in[8];
  const int*   eBA   = (const int*)d_in[9];

  const int NA = in_sizes[0] / TD;
  const int NB = in_sizes[1] / TD;
  const int E  = in_sizes[2];
  const long long nAe = (long long)NA * TD;
  const long long nBe = (long long)NB * TD;

  // workspace layout: gsum[128] | alpha[1] (padded) | curA | curB | accA | accB
  float* wsf     = (float*)d_ws;
  float* gsum    = wsf;
  float* alpha_p = wsf + 128;
  float* curA    = wsf + 256;
  float* curB    = curA + nAe;
  float* accA    = curB + nBe;
  float* accB    = accA + nAe;

  float* outA = (float*)d_out;
  float* outB = outA + nAe;

  k_zero128<<<1, 128, 0, stream>>>(gsum);
  k_colsum<<<512, 256, 0, stream>>>(xA, xB, NA, NB, gsum);
  k_alpha<<<1, 256, 0, stream>>>(gsum, fc1_w, fc1_b, fc2_w, fc2_b, alpha_p);

  k_prep<<<2048, 256, 0, stream>>>(xA, curA, accA, alpha_p, nAe);
  k_prep<<<2048, 256, 0, stream>>>(xB, curB, accB, alpha_p, nBe);

  const int ewBlocks = (E + 7) / 8;     // 8 edges (waves) per 256-thread block
  for (int hop = 0; hop < 3; ++hop) {
    // simultaneous update: both scatters read prev cur before any advance
    k_scatter<<<ewBlocks, 256, 0, stream>>>(curB, accA, eBA, attBA, alpha_p, E);
    k_scatter<<<ewBlocks, 256, 0, stream>>>(curA, accB, eAB, attAB, alpha_p, E);
    if (hop < 2) {
      k_advance<<<2048, 256, 0, stream>>>(xA, curA, accA, alpha_p, nAe);
      k_advance<<<2048, 256, 0, stream>>>(xB, curB, accB, alpha_p, nBe);
    } else {
      k_final<<<2048, 256, 0, stream>>>(accA, outA, nAe);
      k_final<<<2048, 256, 0, stream>>>(accB, outB, nBe);
    }
  }
}
